// AttentionBlock_5806795784209
// MI455X (gfx1250) — compile-verified, matrix operand order checked by reading, not run on hardware
//
#include <hip/hip_runtime.h>

// ---------------------------------------------------------------------------
// CDNA5 / gfx1250 WMMA implementation of the distance-attention block.
// wave32, v_wmma_f32_16x16x32_f16 everywhere, flash attention, f16 QKV,
// async global->LDS staging (ASYNCcnt) for attention Q/K tiles,
// vectorized (b128) staging for everything else.
// ---------------------------------------------------------------------------

typedef __attribute__((ext_vector_type(16))) _Float16 v16h;
typedef __attribute__((ext_vector_type(8)))  _Float16 v8h;
typedef __attribute__((ext_vector_type(4)))  _Float16 v4h;
typedef __attribute__((ext_vector_type(8)))  float    v8f;

#define B_     8
#define N_     1024
#define D_     256
#define H_     8
#define DH_    64
#define DE_    1024
#define DENC_  32
#define DQ_    512
#define DV_    2048
#define DQKV_  3072
#define MROWS  (B_ * N_)   // 8192

__device__ __forceinline__ float silu_f(float x) { return x / (1.0f + __expf(-x)); }

__device__ __forceinline__ v8f wmma_f16(v16h a, v16h b, v8f c) {
  return __builtin_amdgcn_wmma_f32_16x16x32_f16(false, a, false, b, (short)0, c,
                                                false, false);
}

// Fragment load per ISA 7.12.2 16-bit layout: lane needs two contiguous
// 16-byte runs of K: [8*hi, 8*hi+8) and [16+8*hi, 16+8*hi+8).
__device__ __forceinline__ v16h load_frag(const _Float16* row, int hi) {
  v8h a = *(const v8h*)(row + 8 * hi);
  v8h b = *(const v8h*)(row + 16 + 8 * hi);
  return __builtin_shufflevector(a, b, 0, 1, 2, 3, 4, 5, 6, 7,
                                 8, 9, 10, 11, 12, 13, 14, 15);
}

__device__ __forceinline__ v4h cvt4(float4 f) {
  v4h h;
  h[0] = (_Float16)f.x; h[1] = (_Float16)f.y;
  h[2] = (_Float16)f.z; h[3] = (_Float16)f.w;
  return h;
}

// ---------------------------------------------------------------------------
// Kernel 1: concat [x | sinusoid-pe] -> xcat (8192 x 288), float4 copies
// ---------------------------------------------------------------------------
__global__ void concat_pe_kernel(const float* __restrict__ x,
                                 const float* __restrict__ pe,
                                 float* __restrict__ xcat) {
  const int KC = D_ + DENC_;
  const int KC4 = KC / 4;                       // 72 float4 per row
  int idx = blockIdx.x * blockDim.x + threadIdx.x;
  const int total = MROWS * KC4;
  for (; idx < total; idx += gridDim.x * blockDim.x) {
    int row = idx / KC4;
    int c4  = (idx - row * KC4) * 4;
    int n   = row & (N_ - 1);
    float4 v = (c4 < D_) ? *(const float4*)&x[(size_t)row * D_ + c4]
                         : *(const float4*)&pe[n * DENC_ + (c4 - D_)];
    *(float4*)&xcat[(size_t)row * KC + c4] = v;
  }
}

// ---------------------------------------------------------------------------
// Kernel 2: FeatureNorm over the SEQUENCE axis + SiLU (thread per feature)
// ---------------------------------------------------------------------------
__global__ __launch_bounds__(256)
void featnorm_silu_kernel(const float* __restrict__ in,
                          const float* __restrict__ g,
                          const float* __restrict__ b,
                          float* __restrict__ out) {
  const int bb = blockIdx.x;
  const int d  = threadIdx.x;               // 256 == D_
  const float* base = in  + (size_t)bb * N_ * D_;
  float*       ob   = out + (size_t)bb * N_ * D_;
  float s = 0.f, s2 = 0.f;
  for (int n = 0; n < N_; ++n) {
    float v = base[(size_t)n * D_ + d];
    s += v; s2 += v * v;
  }
  const float mean = s * (1.0f / N_);
  const float var  = s2 * (1.0f / N_) - mean * mean;   // biased variance
  const float inv  = rsqrtf(var + 1e-5f);
  const float gg = g[d], bv = b[d];
  for (int n = 0; n < N_; ++n) {
    float v = base[(size_t)n * D_ + d];
    ob[(size_t)n * D_ + d] = silu_f(gg * (v - mean) * inv + bv);
  }
}

// ---------------------------------------------------------------------------
// Kernel 3: generic tiled WMMA GEMM  C = act(A[M,K] @ W[K,N] + bias) (+ R)
// Block tile 64x128, BK=32, 8 waves (2x4), 2x2 WMMA per wave.
// A staged row-major, B staged TRANSPOSED; fragments are 2x ds_load_b128.
// ---------------------------------------------------------------------------
template <int ACT, typename OT>
__global__ __launch_bounds__(256)
void gemm_wmma_kernel(const float* __restrict__ A, const float* __restrict__ W,
                      const float* __restrict__ bias, const float* __restrict__ R,
                      OT* __restrict__ C, int M, int Nn, int K) {
  __shared__ __align__(16) _Float16 As[64][40];    // [row][k]  (80B rows)
  __shared__ __align__(16) _Float16 Bst[128][40];  // [col][k]  transposed

  const int tid  = threadIdx.x;
  const int lane = tid & 31;
  const int wave = tid >> 5;
  const int wm   = wave >> 2;        // 0..1 : 32-row band
  const int wn   = wave & 3;         // 0..3 : 32-col band
  const int lm   = lane & 15;
  const int hi   = (lane & 16) ? 1 : 0;
  const int row0 = blockIdx.y * 64;
  const int col0 = blockIdx.x * 128;

  v8f acc[2][2] = {};

  for (int k0 = 0; k0 < K; k0 += 32) {
    // A tile: 64 x 32 f32 -> 512 float4 groups, b128 load + b64 ds store
    for (int g = tid; g < 64 * 8; g += 256) {
      int r = g >> 3, c4 = (g & 7) * 4;
      float4 f = *(const float4*)&A[(size_t)(row0 + r) * K + k0 + c4];
      *(v4h*)&As[r][c4] = cvt4(f);
    }
    // B tile: 32 x 128 f32, coalesced float4 over cols, transposed scatter
    for (int g = tid; g < 32 * 32; g += 256) {
      int r = g >> 5, c4 = (g & 31) * 4;
      float4 f = *(const float4*)&W[(size_t)(k0 + r) * Nn + col0 + c4];
      v4h h = cvt4(f);
      Bst[c4 + 0][r] = h[0];
      Bst[c4 + 1][r] = h[1];
      Bst[c4 + 2][r] = h[2];
      Bst[c4 + 3][r] = h[3];
    }
    if (k0 + 32 < K) {   // global_prefetch_b8 hints for next tiles
      __builtin_prefetch(&A[(size_t)(row0 + (tid >> 2)) * K + k0 + 32], 0, 1);
      __builtin_prefetch(&W[(size_t)(k0 + 32 + (tid >> 4)) * Nn + col0 + (tid & 15) * 8], 0, 1);
    }
    __syncthreads();

    v16h af[2], bf[2];
    for (int mt = 0; mt < 2; ++mt)
      af[mt] = load_frag(&As[wm * 32 + mt * 16 + lm][0], hi);
    for (int nt = 0; nt < 2; ++nt)
      bf[nt] = load_frag(&Bst[wn * 32 + nt * 16 + lm][0], hi);
    for (int mt = 0; mt < 2; ++mt)
      for (int nt = 0; nt < 2; ++nt)
        acc[mt][nt] = wmma_f16(af[mt], bf[nt], acc[mt][nt]);
    __syncthreads();
  }

  const int hi8 = hi * 8;
  for (int mt = 0; mt < 2; ++mt)
    for (int nt = 0; nt < 2; ++nt)
      for (int r = 0; r < 8; ++r) {
        int grow = row0 + wm * 32 + mt * 16 + r + hi8;
        int gcol = col0 + wn * 32 + nt * 16 + lm;
        float v = acc[mt][nt][r] + bias[gcol];
        if (ACT == 1) v = silu_f(v);
        if (R) v += R[(size_t)grow * Nn + gcol];
        C[(size_t)grow * Nn + gcol] = (OT)v;
      }
}

// ---------------------------------------------------------------------------
// Kernel 4: flash-style distance attention on f16 QKV.
//   S = -max(|q|^2 + |k|^2 - 2 qk^T, 0) / scale^2 ; online softmax ; O = P V
// Grid (N/32, H, B); 8 waves split 2 (16-query tile) x 4 (64-wide V chunk).
// Q/K tiles staged with global_load_async_to_lds_b128 (ASYNCcnt).
// ---------------------------------------------------------------------------
__global__ __launch_bounds__(256)
void attn_wmma_kernel(const _Float16* __restrict__ qkv,
                      const float* __restrict__ scale_p,
                      float* __restrict__ out) {
  __shared__ __align__(16) _Float16 Qs[32][72];    // [q][dh]   144B rows
  __shared__ __align__(16) _Float16 Ks[32][72];    // [k][dh]
  __shared__ __align__(16) _Float16 Vst[256][40];  // [dim][k]  transposed
  __shared__ __align__(16) _Float16 Ps[2][16][40]; // [mtile][q][k]
  __shared__ float q2s[32], k2s[32];

  const int tid = threadIdx.x, lane = tid & 31, wave = tid >> 5;
  const int wm = wave & 1;               // 16-query tile
  const int wn = wave >> 1;              // 64-wide value chunk
  const int lm = lane & 15;
  const int hi = (lane & 16) ? 1 : 0;
  const int hi8 = hi * 8;
  const int qt = blockIdx.x, hh = blockIdx.y, bb = blockIdx.z;

  const float sc = scale_p[0];
  const float inv_s2 = 1.0f / (sc * sc);

  const size_t rowbase = (size_t)bb * N_;
  const _Float16* qkb = qkv + rowbase * DQKV_;

  // --- stage Q tile (32 x 64 f16 = 32 rows x 128B): one async b128 / thread
  {
    int r = tid >> 3, s8 = (tid & 7) * 8;
    const _Float16* gp = qkb + (size_t)(qt * 32 + r) * DQKV_ + hh * DH_ + s8;
    unsigned ldso = (unsigned)(uintptr_t)&Qs[r][s8];
    asm volatile("global_load_async_to_lds_b128 %0, %1, off"
                 :: "v"(ldso), "v"(gp) : "memory");
  }
  if (tid < 32) {
    const v8h* qp = (const v8h*)(qkb + (size_t)(qt * 32 + tid) * DQKV_ + hh * DH_);
    float s = 0.f;
    for (int c = 0; c < 8; ++c) {
      v8h v = qp[c];
      for (int j = 0; j < 8; ++j) { float f = (float)v[j]; s += f * f; }
    }
    q2s[tid] = s;
  }

  v8f o[4] = {};
  float m_acc[8], l_acc[8];
  for (int r = 0; r < 8; ++r) { m_acc[r] = -1e30f; l_acc[r] = 0.f; }

  for (int kt = 0; kt < N_ / 32; ++kt) {
    asm volatile("s_wait_asynccnt 0x0" ::: "memory");
    __syncthreads();   // protect Ks/Vst/Ps from previous iteration

    // --- K tile async (32 x 64 f16): one b128 / thread
    {
      int r = tid >> 3, s8 = (tid & 7) * 8;
      const _Float16* gp =
          qkb + (size_t)(kt * 32 + r) * DQKV_ + DQ_ + hh * DH_ + s8;
      unsigned ldso = (unsigned)(uintptr_t)&Ks[r][s8];
      asm volatile("global_load_async_to_lds_b128 %0, %1, off"
                   :: "v"(ldso), "v"(gp) : "memory");
    }
    // --- V tile: b128 loads (8 halves), transposed 8-way scatter into Vst
    for (int g = tid; g < 32 * 32; g += 256) {
      int r = g >> 5, c8 = (g & 31) * 8;               // coalesced over c8
      v8h v = *(const v8h*)&qkb[(size_t)(kt * 32 + r) * DQKV_ +
                                2 * DQ_ + hh * D_ + c8];
      for (int j = 0; j < 8; ++j) Vst[c8 + j][r] = v[j];
    }
    if (tid < 32) {
      const v8h* kp =
          (const v8h*)(qkb + (size_t)(kt * 32 + tid) * DQKV_ + DQ_ + hh * DH_);
      float s = 0.f;
      for (int c = 0; c < 8; ++c) {
        v8h v = kp[c];
        for (int j = 0; j < 8; ++j) { float f = (float)v[j]; s += f * f; }
      }
      k2s[tid] = s;
    }
    asm volatile("s_wait_asynccnt 0x0" ::: "memory");
    __syncthreads();

    // --- S tile: 16 queries x 32 keys, K-dim = DH = 64 (2 wmma steps/tile)
    v8f s0 = {}, s1 = {};
    for (int kk = 0; kk < DH_; kk += 32) {
      v16h aq = load_frag(&Qs[wm * 16 + lm][kk], hi);
      v16h b0 = load_frag(&Ks[lm][kk], hi);
      v16h b1 = load_frag(&Ks[16 + lm][kk], hi);
      s0 = wmma_f16(aq, b0, s0);
      s1 = wmma_f16(aq, b1, s1);
    }

    // --- scores + online softmax (C layout: row = r + 8*hi, col = lm)
    for (int r = 0; r < 8; ++r) {
      int qrow = wm * 16 + r + hi8;
      float d0 = fmaxf(q2s[qrow] + k2s[lm]      - 2.f * s0[r], 0.f);
      float d1 = fmaxf(q2s[qrow] + k2s[16 + lm] - 2.f * s1[r], 0.f);
      float t0 = -d0 * inv_s2, t1 = -d1 * inv_s2;
      float mx = fmaxf(t0, t1);
      for (int m = 1; m < 16; m <<= 1) mx = fmaxf(mx, __shfl_xor(mx, m, 32));
      float mnew = fmaxf(m_acc[r], mx);
      float corr = __expf(m_acc[r] - mnew);
      float e0 = __expf(t0 - mnew), e1 = __expf(t1 - mnew);
      float rs = e0 + e1;
      for (int m = 1; m < 16; m <<= 1) rs += __shfl_xor(rs, m, 32);
      l_acc[r] = l_acc[r] * corr + rs;
      m_acc[r] = mnew;
      for (int nt = 0; nt < 4; ++nt) o[nt][r] *= corr;
      if (wn == 0) {   // transpose P into A-fragment layout via LDS
        Ps[wm][r + hi8][lm]      = (_Float16)e0;
        Ps[wm][r + hi8][16 + lm] = (_Float16)e1;
      }
    }
    __syncthreads();

    // --- O += P (16x32) @ V (32 x 64-chunk): 4 WMMAs / wave
    v16h pf = load_frag(&Ps[wm][lm][0], hi);
    for (int nt = 0; nt < 4; ++nt) {
      v16h vf = load_frag(&Vst[wn * 64 + nt * 16 + lm][0], hi);
      o[nt] = wmma_f16(pf, vf, o[nt]);
    }
  }

  // normalize and write (B, N, H*256) for the merge GEMM
  for (int nt = 0; nt < 4; ++nt)
    for (int r = 0; r < 8; ++r) {
      int q = qt * 32 + wm * 16 + r + hi8;
      float inv_l = 1.0f / l_acc[r];
      out[(rowbase + q) * (size_t)DV_ + hh * D_ + wn * 64 + nt * 16 + lm] =
          o[nt][r] * inv_l;
    }
}

// ---------------------------------------------------------------------------
extern "C" void kernel_launch(void* const* d_in, const int* in_sizes, int n_in,
                              void* d_out, int out_size, void* d_ws, size_t ws_size,
                              hipStream_t stream) {
  (void)in_sizes; (void)n_in; (void)out_size; (void)ws_size;
  const float* x       = (const float*)d_in[0];
  const float* pe      = (const float*)d_in[1];
  const float* pe_w    = (const float*)d_in[2];
  const float* pe_b    = (const float*)d_in[3];
  const float* fn1_g   = (const float*)d_in[4];
  const float* fn1_b   = (const float*)d_in[5];
  const float* qkv_w   = (const float*)d_in[6];
  const float* qkv_b   = (const float*)d_in[7];
  const float* scale   = (const float*)d_in[8];
  const float* merge_w = (const float*)d_in[9];
  const float* merge_b = (const float*)d_in[10];
  const float* fn2_g   = (const float*)d_in[11];
  const float* fn2_b   = (const float*)d_in[12];
  const float* ff1_w   = (const float*)d_in[13];
  const float* ff1_b   = (const float*)d_in[14];
  const float* ff2_w   = (const float*)d_in[15];
  const float* ff2_b   = (const float*)d_in[16];
  float* out = (float*)d_out;

  float* ws = (float*)d_ws;
  size_t off = 0;
  float* xcat = ws + off; off += (size_t)MROWS * (D_ + DENC_);
  float* h    = ws + off; off += (size_t)MROWS * D_;
  float* hact = ws + off; off += (size_t)MROWS * D_;
  float* qkvr = ws + off; off += (size_t)MROWS * DQKV_;  // f16 qkv / f32 f1
  float* aout = ws + off; off += (size_t)MROWS * DV_;
  _Float16* qkvh = (_Float16*)qkvr;
  float* xr   = xcat;   // xcat dead after step 2
  float* xact = h;      // h dead after step 3
  float* f1   = qkvr;   // qkv dead after attention

  // 1) xcat = [x | pe]
  concat_pe_kernel<<<1024, 256, 0, stream>>>(x, pe, xcat);
  // 2) h = xcat @ pe_w + pe_b            (8192x256, K=288)
  gemm_wmma_kernel<0, float><<<dim3(D_ / 128, MROWS / 64), 256, 0, stream>>>(
      xcat, pe_w, pe_b, nullptr, h, MROWS, D_, D_ + DENC_);
  // 3) hact = silu(featnorm_seq(h))
  featnorm_silu_kernel<<<B_, 256, 0, stream>>>(h, fn1_g, fn1_b, hact);
  // 4) qkv(f16) = hact @ qkv_w + qkv_b   (8192x3072, K=256)
  gemm_wmma_kernel<0, _Float16><<<dim3(DQKV_ / 128, MROWS / 64), 256, 0, stream>>>(
      hact, qkv_w, qkv_b, nullptr, qkvh, MROWS, DQKV_, D_);
  // 5) flash distance-attention -> aout  (8192x2048)
  attn_wmma_kernel<<<dim3(N_ / 32, H_, B_), 256, 0, stream>>>(qkvh, scale, aout);
  // 6) xr = x + aout @ merge_w + merge_b (8192x256, K=2048)
  gemm_wmma_kernel<0, float><<<dim3(D_ / 128, MROWS / 64), 256, 0, stream>>>(
      aout, merge_w, merge_b, x, xr, MROWS, D_, DV_);
  // 7) xact = silu(featnorm_seq(xr))
  featnorm_silu_kernel<<<B_, 256, 0, stream>>>(xr, fn2_g, fn2_b, xact);
  // 8) f1 = silu(xact @ ff1_w + ff1_b)   (8192x1024, K=256)
  gemm_wmma_kernel<1, float><<<dim3(DE_ / 128, MROWS / 64), 256, 0, stream>>>(
      xact, ff1_w, ff1_b, nullptr, f1, MROWS, DE_, D_);
  // 9) out = f1 @ ff2_w + ff2_b + xr     (8192x256, K=1024)
  gemm_wmma_kernel<0, float><<<dim3(D_ / 128, MROWS / 64), 256, 0, stream>>>(
      f1, ff2_w, ff2_b, xr, out, MROWS, D_, DE_);
}
